// RoPEMultiheadSelfAttention_12128987644259
// MI455X (gfx1250) — matrix operand order repaired and checked by reading, not compiled, not run on hardware
//
#include <hip/hip_runtime.h>

// RoPE multi-head self-attention for gfx1250 (MI455X).
// All matmuls on v_wmma_f32_16x16x32_f16 (fp32 accum). V tiles in the
// attention kernel are staged to LDS with global_load_async_to_lds_b128
// (ASYNCcnt) overlapped with the softmax compute.
//
// Workspace layout (_Float16), total 48 MB:
//   q    : [B][H][L][64]    NELEM
//   k    : [B][H][L][64]    NELEM
//   v    : [B][H][L][64]    NELEM
//   ctx  : [B][L][1024]     NELEM
//   xh   : [B][L][1024]     NELEM        (f16 copy of x)
//   wqh  : [3072][1024]     3*D*D        (f16 copy of W_qkv)
//   woh  : [1024][1024]     D*D          (f16 copy of W_out)

#define B_ 2
#define L_ 2048
#define D_ 1024
#define H_ 16
#define DH_ 64
#define NELEM ((size_t)B_ * H_ * L_ * DH_) // 4,194,304

typedef __attribute__((ext_vector_type(16))) _Float16 v16h;
typedef __attribute__((ext_vector_type(8)))  _Float16 v8h;
typedef __attribute__((ext_vector_type(4)))  _Float16 v4h;
typedef __attribute__((ext_vector_type(8)))  float    v8f;

__device__ inline v8f wmma_f16(v16h a, v16h b, v8f c) {
  return __builtin_amdgcn_wmma_f32_16x16x32_f16(
      false, a, false, b, (short)0, c, false, false);
}

// A-fragment (16x32 f16). Lane holds row (lane&15);
// lanes 0-15: K = 0..15, lanes 16-31: K = 16..31.
__device__ inline v16h load_a_f16(const _Float16* row, int k0, int lane) {
  const int kb = k0 + ((lane & 16) ? 16 : 0);
  v8h lo = *(const v8h*)(row + kb);
  v8h hi = *(const v8h*)(row + kb + 8);
  return __builtin_shufflevector(lo, hi, 0,1,2,3,4,5,6,7,8,9,10,11,12,13,14,15);
}

// B-fragment (32x16 f16). Lane holds column (lane&15);
// lanes 0-15: K = 0..15, lanes 16-31: K = 16..31 (contiguous in the W row).
__device__ inline v16h load_b_f16(const _Float16* row, int k0, int lane) {
  const int kb = k0 + ((lane & 16) ? 16 : 0);
  v8h lo = *(const v8h*)(row + kb);
  v8h hi = *(const v8h*)(row + kb + 8);
  return __builtin_shufflevector(lo, hi, 0,1,2,3,4,5,6,7,8,9,10,11,12,13,14,15);
}

// ---------------------------------------------------------------------------
// Kernel 0: f32 -> f16 bulk convert (vectorized x4). One-shot; removes all
// v_cvt traffic from the GEMM hot loops.
// ---------------------------------------------------------------------------
__global__ __launch_bounds__(256) void cvt_f16_kernel(
    const float* __restrict__ in, _Float16* __restrict__ out, int n4)
{
  const int i = blockIdx.x * 256 + threadIdx.x;
  if (i < n4) {
    float4 f = ((const float4*)in)[i];
    v4h h;
    h[0] = (_Float16)f.x; h[1] = (_Float16)f.y;
    h[2] = (_Float16)f.z; h[3] = (_Float16)f.w;
    ((v4h*)out)[i] = h;
  }
}

// ---------------------------------------------------------------------------
// Kernel 1: qkv = x @ W_qkv^T (f16 operands), fused RoPE on q/k.
// One wave32 computes a 32(L) x 64(Dh) tile: 2 A-frags reuse 4 B-frags.
// Grid: (L/32, B*3*H).
// ---------------------------------------------------------------------------
__global__ __launch_bounds__(32) void qkv_rope_kernel(
    const _Float16* __restrict__ xh, const _Float16* __restrict__ wqh,
    const int* __restrict__ pos_ids,
    _Float16* __restrict__ qb, _Float16* __restrict__ kbuf,
    _Float16* __restrict__ vb)
{
  const int lane = threadIdx.x;
  int y = blockIdx.y;                 // b*48 + s*16 + h
  const int b = y / 48;  y -= b * 48;
  const int s = y >> 4;               // 0=q 1=k 2=v
  const int h = y & 15;
  const int l0 = blockIdx.x * 32;
  const int hi8 = (lane & 16) ? 8 : 0;

  const _Float16* arow0 = xh + (size_t)(b * L_ + l0 +      (lane & 15)) * D_;
  const _Float16* arow1 = xh + (size_t)(b * L_ + l0 + 16 + (lane & 15)) * D_;
  const int ebase = s * D_ + h * DH_;

  v8f acc[2][4] = {{v8f{}, v8f{}, v8f{}, v8f{}}, {v8f{}, v8f{}, v8f{}, v8f{}}};
  for (int k0 = 0; k0 < D_; k0 += 32) {
    __builtin_prefetch((const void*)(arow0 + k0 + 128), 0, 0);
    __builtin_prefetch((const void*)(arow1 + k0 + 128), 0, 0);
    v16h a0 = load_a_f16(arow0, k0, lane);
    v16h a1 = load_a_f16(arow1, k0, lane);
#pragma unroll
    for (int t = 0; t < 4; ++t) {
      const _Float16* wrow = wqh + (size_t)(ebase + 16 * t + (lane & 15)) * D_;
      v16h bf = load_b_f16(wrow, k0, lane);
      acc[0][t] = wmma_f16(a0, bf, acc[0][t]);
      acc[1][t] = wmma_f16(a1, bf, acc[1][t]);
    }
  }

#pragma unroll
  for (int mt = 0; mt < 2; ++mt) {
    if (s == 2) { // V: no rope
#pragma unroll
      for (int r = 0; r < 8; ++r) {
        const int l = l0 + 16 * mt + r + hi8;
        _Float16* dst = vb + (((size_t)(b * H_ + h)) * L_ + l) * DH_;
#pragma unroll
        for (int t = 0; t < 4; ++t)
          dst[16 * t + (lane & 15)] = (_Float16)acc[mt][t][r];
      }
    } else {     // Q/K: interleaved RoPE; pair partner sits in adjacent lane
      _Float16* outp = (s == 0) ? qb : kbuf;
#pragma unroll
      for (int r = 0; r < 8; ++r) {
        const int l = l0 + 16 * mt + r + hi8;
        const float pos = (float)pos_ids[b * L_ + l];
        _Float16* dst = outp + (((size_t)(b * H_ + h)) * L_ + l) * DH_;
#pragma unroll
        for (int t = 0; t < 4; ++t) {
          const int dh = 16 * t + (lane & 15);
          const float val   = acc[mt][t][r];
          const float other = __shfl_xor(val, 1);
          const float f = pos * __expf(-(float)(dh & ~1) * (9.210340371976184f / 64.0f));
          const float c = __cosf(f), sn = __sinf(f);
          const float o = (lane & 1) ? (other * sn + val * c)
                                     : (val * c - other * sn);
          dst[dh] = (_Float16)o;
        }
      }
    }
  }
}

// ---------------------------------------------------------------------------
// Kernel 2: causal flash attention per (b,h,16-row tile). Key blocks of 32.
// V tile is DMA'd to LDS with global_load_async_to_lds_b128, overlapped with
// the QK^T + softmax work; rows padded to 66 f16 for bank-conflict-free use.
// Grid: (L/16, B*H), 1 wave32.
// ---------------------------------------------------------------------------
__global__ __launch_bounds__(32) void attn_kernel(
    const _Float16* __restrict__ qb, const _Float16* __restrict__ kbuf,
    const _Float16* __restrict__ vb, _Float16* __restrict__ ctx)
{
  __shared__ float    pS[16 * 32];  // P tile: C-layout -> A-layout staging
  __shared__ _Float16 vS[32 * 66];  // 32 key rows x 64 dh, stride 66 (pad)

  const int lane = threadIdx.x;
  const int y = blockIdx.y;
  const int b = y >> 4, h = y & 15;
  const int l0 = blockIdx.x * 16;
  const int hi8 = (lane & 16) ? 8 : 0;
  const size_t headOff = ((size_t)(b * H_ + h)) * L_ * DH_;
  const unsigned ldsV = (unsigned)(unsigned long long)(&vS[0]); // LDS byte offset

  const _Float16* qrow = qb + headOff + (size_t)(l0 + (lane & 15)) * DH_;
  const v16h aq0 = load_a_f16(qrow, 0, lane);
  const v16h aq1 = load_a_f16(qrow, 32, lane);

  float mrun[8], srun[8];
#pragma unroll
  for (int r = 0; r < 8; ++r) { mrun[r] = -1e30f; srun[r] = 0.f; }
  v8f acc[4] = {v8f{}, v8f{}, v8f{}, v8f{}};

  const int kend = l0 + 15;
  for (int m0 = 0; m0 <= kend; m0 += 32) {
    // --- kick off async DMA of V[m0..m0+31][0..63] into LDS ---------------
    {
      // WAR: previous iteration's ds reads of vS must retire first.
      asm volatile("s_wait_dscnt 0x0" ::: "memory");
      const _Float16* gsrc = vb + headOff + (size_t)(m0 + lane) * DH_;
      const unsigned ldst = ldsV + (unsigned)lane * (66u * 2u);
#pragma unroll
      for (int j = 0; j < 8; ++j) {
        const unsigned la = ldst + 16u * (unsigned)j;
        const unsigned long long ga = (unsigned long long)(gsrc + 8 * j);
        asm volatile("global_load_async_to_lds_b128 %0, %1, off"
                     :: "v"(la), "v"(ga) : "memory");
      }
    }

    // --- scores: 16 rows x 32 keys (overlaps with the DMA) ----------------
    v8f sc[2];
#pragma unroll
    for (int u = 0; u < 2; ++u) {
      const _Float16* krow =
          kbuf + headOff + (size_t)(m0 + 16 * u + (lane & 15)) * DH_;
      v16h bk0 = load_b_f16(krow, 0, lane);
      v16h bk1 = load_b_f16(krow, 32, lane);
      v8f sa = {};
      sa = wmma_f16(aq0, bk0, sa);
      sa = wmma_f16(aq1, bk1, sa);
      sc[u] = sa;
    }
    // --- scale + causal mask ---------------------------------------------
#pragma unroll
    for (int u = 0; u < 2; ++u) {
      const int col = m0 + 16 * u + (lane & 15);
#pragma unroll
      for (int r = 0; r < 8; ++r) {
        const int row = l0 + r + hi8;
        const float v = sc[u][r] * 0.125f; // 1/sqrt(64)
        sc[u][r] = (col <= row) ? v : -1e30f;
      }
    }
    // --- online softmax (row stats reduce within each 16-lane half) ------
#pragma unroll
    for (int r = 0; r < 8; ++r) {
      float m = fmaxf(sc[0][r], sc[1][r]);
      m = fmaxf(m, __shfl_xor(m, 1));
      m = fmaxf(m, __shfl_xor(m, 2));
      m = fmaxf(m, __shfl_xor(m, 4));
      m = fmaxf(m, __shfl_xor(m, 8));
      const float mnew = fmaxf(mrun[r], m);
      const float resc = __expf(mrun[r] - mnew);
      const float p0 = __expf(sc[0][r] - mnew);
      const float p1 = __expf(sc[1][r] - mnew);
      float ps = p0 + p1;
      ps += __shfl_xor(ps, 1);
      ps += __shfl_xor(ps, 2);
      ps += __shfl_xor(ps, 4);
      ps += __shfl_xor(ps, 8);
      srun[r] = srun[r] * resc + ps;
      mrun[r] = mnew;
#pragma unroll
      for (int t = 0; t < 4; ++t) acc[t][r] *= resc;
      pS[(r + hi8) * 32 + (lane & 15)]      = p0;
      pS[(r + hi8) * 32 + 16 + (lane & 15)] = p1;
    }
    __syncthreads();
    // --- re-layout P to A-fragment ---------------------------------------
    v16h pa;
    {
      const int m = lane & 15;
#pragma unroll
      for (int i = 0; i < 16; ++i) {
        const int kidx = i + ((lane & 16) ? 16 : 0);
        pa[i] = (_Float16)pS[m * 32 + kidx];
      }
    }
    __syncthreads();
    // --- wait for the V DMA, then ctx += P @ V ----------------------------
    asm volatile("s_wait_asynccnt 0x0" ::: "memory");
    const int kofs = (lane & 16) ? 16 : 0;
#pragma unroll
    for (int t = 0; t < 4; ++t) {
      v16h bv;
#pragma unroll
      for (int i = 0; i < 16; ++i)
        bv[i] = vS[(kofs + i) * 66 + 16 * t + (lane & 15)];
      acc[t] = wmma_f16(pa, bv, acc[t]);
    }
  }

  // epilogue: normalize, write ctx as [B][L][D] with col = h*64+dh
#pragma unroll
  for (int r = 0; r < 8; ++r) {
    const float inv = 1.0f / srun[r];
    const int l = l0 + r + hi8;
    _Float16* dst = ctx + ((size_t)(b * L_) + l) * D_ + h * DH_;
#pragma unroll
    for (int t = 0; t < 4; ++t)
      dst[16 * t + (lane & 15)] = (_Float16)(acc[t][r] * inv);
  }
}

// ---------------------------------------------------------------------------
// Kernel 3: out = ctx @ W_out^T (f16 operands, fp32 out). 32x64 tile/wave.
// Grid: (L/32, B*16).
// ---------------------------------------------------------------------------
__global__ __launch_bounds__(32) void outproj_kernel(
    const _Float16* __restrict__ ctx, const _Float16* __restrict__ woh,
    float* __restrict__ out)
{
  const int lane = threadIdx.x;
  const int y = blockIdx.y;
  const int b = y >> 4, n64 = y & 15;
  const int l0 = blockIdx.x * 32;
  const int hi8 = (lane & 16) ? 8 : 0;

  const _Float16* arow0 = ctx + (size_t)(b * L_ + l0 +      (lane & 15)) * D_;
  const _Float16* arow1 = ctx + (size_t)(b * L_ + l0 + 16 + (lane & 15)) * D_;
  v8f acc[2][4] = {{v8f{}, v8f{}, v8f{}, v8f{}}, {v8f{}, v8f{}, v8f{}, v8f{}}};
  for (int k0 = 0; k0 < D_; k0 += 32) {
    __builtin_prefetch((const void*)(arow0 + k0 + 128), 0, 0);
    __builtin_prefetch((const void*)(arow1 + k0 + 128), 0, 0);
    v16h a0 = load_a_f16(arow0, k0, lane);
    v16h a1 = load_a_f16(arow1, k0, lane);
#pragma unroll
    for (int t = 0; t < 4; ++t) {
      const _Float16* wrow = woh + (size_t)(n64 * 64 + 16 * t + (lane & 15)) * D_;
      v16h bf = load_b_f16(wrow, k0, lane);
      acc[0][t] = wmma_f16(a0, bf, acc[0][t]);
      acc[1][t] = wmma_f16(a1, bf, acc[1][t]);
    }
  }
#pragma unroll
  for (int mt = 0; mt < 2; ++mt) {
#pragma unroll
    for (int r = 0; r < 8; ++r) {
      const int l = l0 + 16 * mt + r + hi8;
      float* dst = out + ((size_t)(b * L_) + l) * D_ + n64 * 64;
#pragma unroll
      for (int t = 0; t < 4; ++t)
        dst[16 * t + (lane & 15)] = acc[mt][t][r];
    }
  }
}

extern "C" void kernel_launch(void* const* d_in, const int* in_sizes, int n_in,
                              void* d_out, int out_size, void* d_ws, size_t ws_size,
                              hipStream_t stream) {
  const float* x    = (const float*)d_in[0];
  const float* Wqkv = (const float*)d_in[1];
  const float* Wout = (const float*)d_in[2];
  const int*   pos  = (const int*)d_in[3];
  // d_in[4] = causal tril mask; implemented analytically in attn_kernel.

  _Float16* ws  = (_Float16*)d_ws;   // needs 48 MB
  _Float16* qb  = ws;
  _Float16* kb  = ws + NELEM;
  _Float16* vb  = ws + 2 * NELEM;
  _Float16* ctx = ws + 3 * NELEM;
  _Float16* xh  = ws + 4 * NELEM;
  _Float16* wqh = xh + (size_t)B_ * L_ * D_;
  _Float16* woh = wqh + (size_t)3 * D_ * D_;
  float* out = (float*)d_out;

  // one-shot f32 -> f16 copies
  {
    const int nx = B_ * L_ * D_ / 4, nq = 3 * D_ * D_ / 4, no = D_ * D_ / 4;
    cvt_f16_kernel<<<(nx + 255) / 256, 256, 0, stream>>>(x, xh, nx);
    cvt_f16_kernel<<<(nq + 255) / 256, 256, 0, stream>>>(Wqkv, wqh, nq);
    cvt_f16_kernel<<<(no + 255) / 256, 256, 0, stream>>>(Wout, woh, no);
  }

  dim3 g1(L_ / 32, B_ * 3 * H_);
  qkv_rope_kernel<<<g1, 32, 0, stream>>>(xh, wqh, pos, qb, kb, vb);

  dim3 g2(L_ / 16, B_ * H_);
  attn_kernel<<<g2, 32, 0, stream>>>(qb, kb, vb, ctx);

  dim3 g3(L_ / 32, B_ * H_);
  outproj_kernel<<<g3, 32, 0, stream>>>(ctx, woh, out);
}
